// InteractionNetwork_45621142618421
// MI455X (gfx1250) — matrix operand order repaired and checked by reading, not compiled, not run on hardware
//
#include <hip/hip_runtime.h>
#include <hip/hip_bf16.h>
#include <stdint.h>

typedef __attribute__((ext_vector_type(16))) __bf16 v16bf;
typedef __attribute__((ext_vector_type(8)))  __bf16 v8bf;
typedef __attribute__((ext_vector_type(4)))  __bf16 v4bf;
typedef __attribute__((ext_vector_type(8)))  float  v8f;
typedef __attribute__((ext_vector_type(4)))  float  v4f;

#define LN_EPS 1e-5f

// ---------------------------------------------------------------------------
// Fused MLP kernel arguments. Each "chunk" is one 128-column slice of the
// concatenated input, with an optional per-row gather index:
//   col_src(row) = cbase[ch] + rowmap(row)*128,
//   rowmap(row)  = cidx[ch] ? cidx[ch][row*cistride+cioff] : row*cistride+cioff
// ---------------------------------------------------------------------------
struct MlpArgs {
  const float* cbase[9];
  const int*   cidx[9];
  int          cistride[9];
  int          cioff[9];
  const __bf16* Wt0; const __bf16* Wt1; const __bf16* Wt2;  // transposed [N][K] bf16
  const float* b0; const float* b1; const float* b2;
  const float* g;  const float* beta;
  const float* res;     // residual source, row-major [nrows][NOUT]
  float*       out;     // output, row-major [nrows][NOUT]
  float*       aggr;    // scatter target (or null), [*][128]
  const int*   sidx;    // receiver index (or null)
  int          sstride; // 1 (edge) or 3 (face-face: one receiver per 128-col group)
  int          nrows;
};

// A fragment: 16x32 bf16 tile in LDS, row stride 128 elements.
// Per ISA 7.12.2: lanes 0-15 hold K 0..7 & 16..23, lanes 16-31 hold K 8..15 & 24..31.
__device__ __forceinline__ v16bf load_a_frag(const __bf16* abase, int lane, int kb) {
  const int m = lane & 15;
  const int h = (lane >> 4) << 3;            // 0 or 8
  const __bf16* p = abase + m * 128 + kb + h;
  v8bf lo = *reinterpret_cast<const v8bf*>(p);       // K = kb+h    .. +7
  v8bf hi = *reinterpret_cast<const v8bf*>(p + 16);  // K = kb+16+h .. +7
  v16bf r;
#pragma unroll
  for (int i = 0; i < 8; ++i) { r[i] = lo[i]; r[8 + i] = hi[i]; }
  return r;
}

// B fragment: weights stored transposed Wt[n][k] (k contiguous).
// Lanes 0-15: column n, K 0..7 & 16..23; lanes 16-31: column n, K 8..15 & 24..31.
__device__ __forceinline__ v16bf load_b_frag(const __bf16* wt, int lane, int K,
                                             int ncol, int kb) {
  const int n = lane & 15;
  const int h = (lane >> 4) << 3;            // 0 or 8
  const __bf16* p = wt + (size_t)(ncol + n) * (size_t)K + kb + h;
  v8bf lo = *reinterpret_cast<const v8bf*>(p);
  v8bf hi = *reinterpret_cast<const v8bf*>(p + 16);
  v16bf r;
#pragma unroll
  for (int i = 0; i < 8; ++i) { r[i] = lo[i]; r[8 + i] = hi[i]; }
  return r;
}

// Vectorized gather of one 16x128 chunk into registers (b128 loads).
// Thread t holds quads i4 = t + k*128, k=0..3: row = i4>>5, col4 = (i4&31)*4.
__device__ __forceinline__ void load_chunk_regs(const MlpArgs& P, int ch, int row0,
                                                int tid, v4f (&regs)[4]) {
  const float* base = P.cbase[ch];
  const int*   idx  = P.cidx[ch];
  const int    istr = P.cistride[ch];
  const int    ioff = P.cioff[ch];
#pragma unroll
  for (int k = 0; k < 4; ++k) {
    const int i4 = tid + k * 128;
    const int r = i4 >> 5, c4 = (i4 & 31) << 2;
    const int row = row0 + r;
    v4f v = {};
    if (row < P.nrows) {
      const long ri = idx ? (long)idx[(long)row * istr + ioff]
                          : ((long)row * istr + ioff);
      v = *reinterpret_cast<const v4f*>(base + ri * 128 + c4);
    }
    regs[k] = v;
  }
}

__device__ __forceinline__ void store_chunk(__bf16* buf, int tid,
                                            const v4f (&regs)[4]) {
#pragma unroll
  for (int k = 0; k < 4; ++k) {
    const int i4 = tid + k * 128;
    const int r = i4 >> 5, c4 = (i4 & 31) << 2;
    // packed v_cvt_pk_bf16_f32 conversion
    *reinterpret_cast<v4bf*>(buf + r * 128 + c4) =
        __builtin_convertvector(regs[k], v4bf);
  }
}

// ---------------------------------------------------------------------------
// Fused: gather -> (Linear,ReLU) -> (Linear,ReLU) -> Linear -> LayerNorm
//        -> residual write + optional atomic scatter-add of the message.
// Block = 128 threads (4 waves), tile = 16 rows. KIN = NCHUNK*128.
// Layer-0 gathers are software-pipelined through a ping-pong LDS buffer.
// ---------------------------------------------------------------------------
template<int NCHUNK, int NOUT>
__global__ __launch_bounds__(128, 2)
void mlp_fused(MlpArgs P) {
  constexpr int KIN = NCHUNK * 128;
  constexpr int T2  = NOUT / 64;             // layer-2 tiles per wave

  __shared__ __align__(16) __bf16 gbufA[16 * 128];
  __shared__ __align__(16) __bf16 gbufB[16 * 128];
  __shared__ __align__(16) __bf16 act0[16 * 128];
  __shared__ __align__(16) __bf16 act1[16 * 128];
  __shared__ __align__(16) float  outF[16 * NOUT];
  __shared__ float mu_s[16], rs_s[16];

  const int tid  = threadIdx.x;
  const int lane = tid & 31;
  const int wave = tid >> 5;
  const int row0 = blockIdx.x * 16;
  const int cn   = lane & 15;                // C-tile column within tile
  const int mb   = (lane >> 4) << 3;         // C-tile row base (0 or 8)

  // ---------------- layer 0: [KIN] -> 128, ReLU (pipelined) ----------------
  v8f acc0[2] = {};
  v4f regs[4];
  load_chunk_regs(P, 0, row0, tid, regs);
#pragma unroll
  for (int ch = 0; ch < NCHUNK; ++ch) {
    __bf16* buf = (ch & 1) ? gbufB : gbufA;
    store_chunk(buf, tid, regs);
    __syncthreads();
    if (ch + 1 < NCHUNK)                      // issue next gather; no wait yet
      load_chunk_regs(P, ch + 1, row0, tid, regs);
#pragma unroll
    for (int kk = 0; kk < 4; ++kk) {
      const v16bf a = load_a_frag(buf, lane, kk * 32);
#pragma unroll
      for (int j = 0; j < 2; ++j) {
        const int ncol = (wave * 2 + j) * 16;
        const v16bf b = load_b_frag(P.Wt0, lane, KIN, ncol, ch * 128 + kk * 32);
        acc0[j] = __builtin_amdgcn_wmma_f32_16x16x32_bf16(
            false, a, false, b, (short)0, acc0[j], false, false);
      }
    }
  }
#pragma unroll
  for (int j = 0; j < 2; ++j) {
    const int ncol = (wave * 2 + j) * 16;
    const float bias = P.b0[ncol + cn];
#pragma unroll
    for (int r = 0; r < 8; ++r) {
      float v = acc0[j][r] + bias;
      v = v > 0.f ? v : 0.f;
      act0[(mb + r) * 128 + ncol + cn] = (__bf16)v;
    }
  }
  __syncthreads();

  // ---------------- layer 1: 128 -> 128, ReLU ----------------
  v8f acc1[2] = {};
#pragma unroll
  for (int kk = 0; kk < 4; ++kk) {
    const v16bf a = load_a_frag(act0, lane, kk * 32);
#pragma unroll
    for (int j = 0; j < 2; ++j) {
      const int ncol = (wave * 2 + j) * 16;
      const v16bf b = load_b_frag(P.Wt1, lane, 128, ncol, kk * 32);
      acc1[j] = __builtin_amdgcn_wmma_f32_16x16x32_bf16(
          false, a, false, b, (short)0, acc1[j], false, false);
    }
  }
#pragma unroll
  for (int j = 0; j < 2; ++j) {
    const int ncol = (wave * 2 + j) * 16;
    const float bias = P.b1[ncol + cn];
#pragma unroll
    for (int r = 0; r < 8; ++r) {
      float v = acc1[j][r] + bias;
      v = v > 0.f ? v : 0.f;
      act1[(mb + r) * 128 + ncol + cn] = (__bf16)v;
    }
  }
  __syncthreads();

  // ---------------- layer 2: 128 -> NOUT (linear) ----------------
  v8f acc2[T2] = {};
#pragma unroll
  for (int kk = 0; kk < 4; ++kk) {
    const v16bf a = load_a_frag(act1, lane, kk * 32);
#pragma unroll
    for (int j = 0; j < T2; ++j) {
      const int ncol = (wave * T2 + j) * 16;
      const v16bf b = load_b_frag(P.Wt2, lane, 128, ncol, kk * 32);
      acc2[j] = __builtin_amdgcn_wmma_f32_16x16x32_bf16(
          false, a, false, b, (short)0, acc2[j], false, false);
    }
  }
#pragma unroll
  for (int j = 0; j < T2; ++j) {
    const int ncol = (wave * T2 + j) * 16;
    const float bias = P.b2[ncol + cn];
#pragma unroll
    for (int r = 0; r < 8; ++r)
      outF[(mb + r) * NOUT + ncol + cn] = acc2[j][r] + bias;
  }
  __syncthreads();

  // ------------- LayerNorm + residual + scatter (vectorized) -------------
  if (tid < 16) {
    const v4f* p = reinterpret_cast<const v4f*>(&outF[tid * NOUT]);
    v4f s = {}, s2 = {};
    for (int c = 0; c < NOUT / 4; ++c) {
      const v4f v = p[c];
      s += v; s2 += v * v;
    }
    const float sum  = s[0] + s[1] + s[2] + s[3];
    const float sum2 = s2[0] + s2[1] + s2[2] + s2[3];
    const float mu = sum / (float)NOUT;
    mu_s[tid] = mu;
    rs_s[tid] = rsqrtf(sum2 / (float)NOUT - mu * mu + LN_EPS);
  }
  __syncthreads();
  for (int i = tid; i < 16 * NOUT / 4; i += 128) {
    const int e = i * 4;
    const int r = e / NOUT, c = e - r * NOUT;
    const int row = row0 + r;
    if (row < P.nrows) {
      const float mu = mu_s[r], rs = rs_s[r];
      const v4f x  = *reinterpret_cast<const v4f*>(&outF[e]);
      const v4f gg = *reinterpret_cast<const v4f*>(P.g + c);
      const v4f bb = *reinterpret_cast<const v4f*>(P.beta + c);
      const v4f y  = (x - mu) * rs * gg + bb;
      const size_t o = (size_t)row * NOUT + c;
      const v4f rr = *reinterpret_cast<const v4f*>(P.res + o);
      *reinterpret_cast<v4f*>(P.out + o) = rr + y;
      if (P.sidx) {                         // c..c+3 stay in one 128-col group
        const int t = P.sidx[(size_t)row * P.sstride + (c >> 7)];
        float* ap = P.aggr + (size_t)t * 128 + (c & 127);
        atomicAdd(ap + 0, y[0]);
        atomicAdd(ap + 1, y[1]);
        atomicAdd(ap + 2, y[2]);
        atomicAdd(ap + 3, y[3]);
      }
    }
  }
}

// Zero the aggregation scratch (b128 stores).
__global__ void k_zero(v4f* p, long n4) {
  const long i = (long)blockIdx.x * blockDim.x + threadIdx.x;
  if (i < n4) p[i] = v4f{};
}

// Convert W[k][n] (f32, row-major [K][N]) -> Wt[n][k] (bf16).
__global__ void k_tconv(const float* __restrict__ W, __bf16* __restrict__ Wt,
                        int K, int N) {
  const long i = (long)blockIdx.x * blockDim.x + threadIdx.x;
  if (i < (long)K * N) {
    const long n = i / K, k = i - n * (long)K;
    Wt[i] = (__bf16)W[k * (long)N + n];
  }
}

// ---------------------------------------------------------------------------
extern "C" void kernel_launch(void* const* d_in, const int* in_sizes, int n_in,
                              void* d_out, int out_size, void* d_ws, size_t ws_size,
                              hipStream_t stream) {
  (void)n_in; (void)out_size; (void)ws_size;
  const int D = 128;
  const float* mesh_n = (const float*)d_in[0];
  const float* obj_n  = (const float*)d_in[1];
  const int* mm_index = (const int*)d_in[2];
  const int* mo_index = (const int*)d_in[3];
  const int* om_index = (const int*)d_in[4];
  const int* ff_index = (const int*)d_in[5];
  const float* e_mm = (const float*)d_in[6];
  const float* e_mo = (const float*)d_in[7];
  const float* e_om = (const float*)d_in[8];
  const float* e_ff = (const float*)d_in[9];

  const int NM  = in_sizes[0] / D;
  const int NO  = in_sizes[1] / D;
  const int EMM = in_sizes[2] / 2;
  const int EMO = in_sizes[3] / 2;
  const int EOM = in_sizes[4] / 2;
  const int EFF = in_sizes[5] / 6;

  const int* mm0 = mm_index, *mm1 = mm_index + EMM;
  const int* mo0 = mo_index, *mo1 = mo_index + EMO;
  const int* om0 = om_index, *om1 = om_index + EOM;
  const int* ff0 = ff_index, *ff1 = ff_index + (size_t)EFF * 3;

  // params flattened depth-first after the 10 data inputs:
  // per MLP: W0,W1,W2,b0,b1,b2,g,beta ; MLP order: mm,mo,om,ff,mesh,obj
  const int pbs[6]  = {10, 18, 26, 34, 42, 50};
  const int kin[6]  = {384, 384, 384, 1152, 512, 256};
  const int nout[6] = {128, 128, 128, 384, 128, 128};

  // Workspace carve: 4 contiguous f32 aggregation buffers, then bf16 weights.
  char* ws = (char*)d_ws;
  float* aggr_mm = (float*)ws; ws += (size_t)NM * D * 4;
  float* aggr_mo = (float*)ws; ws += (size_t)NO * D * 4;
  float* aggr_om = (float*)ws; ws += (size_t)NM * D * 4;
  float* aggr_ff = (float*)ws; ws += (size_t)NM * D * 4;
  ws = (char*)(((uintptr_t)ws + 255) & ~(uintptr_t)255);

  __bf16* Wt[6][3];
  for (int m = 0; m < 6; ++m) {
    const int Ks[3] = {kin[m], 128, 128};
    const int Ns[3] = {128, 128, nout[m]};
    for (int l = 0; l < 3; ++l) {
      Wt[m][l] = (__bf16*)ws;
      const size_t e = (size_t)Ks[l] * Ns[l];
      ws += ((e * 2 + 255) & ~(size_t)255);
    }
  }

  // Zero aggregation buffers (contiguous block starting at aggr_mm).
  {
    const long n4 = ((long)NM * 3 + NO) * D / 4;
    k_zero<<<(unsigned)((n4 + 255) / 256), 256, 0, stream>>>((v4f*)aggr_mm, n4);
  }
  // Convert + transpose weights to bf16 (tiny; L2-resident afterwards).
  for (int m = 0; m < 6; ++m) {
    const int Ks[3] = {kin[m], 128, 128};
    const int Ns[3] = {128, 128, nout[m]};
    for (int l = 0; l < 3; ++l) {
      const long e = (long)Ks[l] * Ns[l];
      k_tconv<<<(unsigned)((e + 255) / 256), 256, 0, stream>>>(
          (const float*)d_in[pbs[m] + l], Wt[m][l], Ks[l], Ns[l]);
    }
  }

  // Output regions, in reference return order.
  float* out_mesh = (float*)d_out;
  float* out_obj  = out_mesh + (size_t)NM * D;
  float* out_emm  = out_obj  + (size_t)NO * D;
  float* out_emo  = out_emm  + (size_t)EMM * D;
  float* out_eom  = out_emo  + (size_t)EMO * D;
  float* out_eff  = out_eom  + (size_t)EOM * D;

  auto set_params = [&](MlpArgs& a, int m) {
    a.Wt0 = Wt[m][0]; a.Wt1 = Wt[m][1]; a.Wt2 = Wt[m][2];
    const int pb = pbs[m];
    a.b0 = (const float*)d_in[pb + 3];
    a.b1 = (const float*)d_in[pb + 4];
    a.b2 = (const float*)d_in[pb + 5];
    a.g  = (const float*)d_in[pb + 6];
    a.beta = (const float*)d_in[pb + 7];
  };
  auto chunk = [&](MlpArgs& a, int ch, const float* base, const int* idx,
                   int istr, int ioff) {
    a.cbase[ch] = base; a.cidx[ch] = idx; a.cistride[ch] = istr; a.cioff[ch] = ioff;
  };

  // ---- edge mm: [e_mm | mesh[mm0] | mesh[mm1]] -> 128, scatter to aggr_mm ----
  { MlpArgs a = {}; set_params(a, 0);
    chunk(a, 0, e_mm,   nullptr, 1, 0);
    chunk(a, 1, mesh_n, mm0,     1, 0);
    chunk(a, 2, mesh_n, mm1,     1, 0);
    a.res = e_mm; a.out = out_emm; a.aggr = aggr_mm; a.sidx = mm1; a.sstride = 1;
    a.nrows = EMM;
    mlp_fused<3, 128><<<(EMM + 15) / 16, 128, 0, stream>>>(a);
  }
  // ---- edge mo: [e_mo | mesh[mo0] | obj[mo1]] -> 128, scatter to aggr_mo ----
  { MlpArgs a = {}; set_params(a, 1);
    chunk(a, 0, e_mo,   nullptr, 1, 0);
    chunk(a, 1, mesh_n, mo0,     1, 0);
    chunk(a, 2, obj_n,  mo1,     1, 0);
    a.res = e_mo; a.out = out_emo; a.aggr = aggr_mo; a.sidx = mo1; a.sstride = 1;
    a.nrows = EMO;
    mlp_fused<3, 128><<<(EMO + 15) / 16, 128, 0, stream>>>(a);
  }
  // ---- edge om: [e_om | obj[om0] | mesh[om1]] -> 128, scatter to aggr_om ----
  { MlpArgs a = {}; set_params(a, 2);
    chunk(a, 0, e_om,   nullptr, 1, 0);
    chunk(a, 1, obj_n,  om0,     1, 0);
    chunk(a, 2, mesh_n, om1,     1, 0);
    a.res = e_om; a.out = out_eom; a.aggr = aggr_om; a.sidx = om1; a.sstride = 1;
    a.nrows = EOM;
    mlp_fused<3, 128><<<(EOM + 15) / 16, 128, 0, stream>>>(a);
  }
  // ---- face-face: per vertex v: [e_ff(v) | mesh[ff0(v)] | mesh[ff1(v)]] -> 384
  { MlpArgs a = {}; set_params(a, 3);
    for (int v = 0; v < 3; ++v) {
      chunk(a, 3 * v + 0, e_ff,   nullptr, 3, v);
      chunk(a, 3 * v + 1, mesh_n, ff0,     3, v);
      chunk(a, 3 * v + 2, mesh_n, ff1,     3, v);
    }
    a.res = e_ff; a.out = out_eff; a.aggr = aggr_ff; a.sidx = ff1; a.sstride = 3;
    a.nrows = EFF;
    mlp_fused<9, 384><<<(EFF + 15) / 16, 128, 0, stream>>>(a);
  }
  // ---- node mesh: [mesh_n | aggr_om | aggr_mm | aggr_ff] -> 128 ----
  { MlpArgs a = {}; set_params(a, 4);
    chunk(a, 0, mesh_n,  nullptr, 1, 0);
    chunk(a, 1, aggr_om, nullptr, 1, 0);
    chunk(a, 2, aggr_mm, nullptr, 1, 0);
    chunk(a, 3, aggr_ff, nullptr, 1, 0);
    a.res = mesh_n; a.out = out_mesh; a.aggr = nullptr; a.sidx = nullptr;
    a.sstride = 1; a.nrows = NM;
    mlp_fused<4, 128><<<(NM + 15) / 16, 128, 0, stream>>>(a);
  }
  // ---- node obj: [obj_n | aggr_mo] -> 128 ----
  { MlpArgs a = {}; set_params(a, 5);
    chunk(a, 0, obj_n,   nullptr, 1, 0);
    chunk(a, 1, aggr_mo, nullptr, 1, 0);
    a.res = obj_n; a.out = out_obj; a.aggr = nullptr; a.sidx = nullptr;
    a.sstride = 1; a.nrows = NO;
    mlp_fused<2, 128><<<(NO + 15) / 16, 128, 0, stream>>>(a);
  }
}
